// AttnWindowPool_45217415693086
// MI455X (gfx1250) — matrix operand order repaired and checked by reading, not compiled, not run on hardware
//
#include <hip/hip_runtime.h>
#include <hip/hip_bf16.h>

// ---------------------------------------------------------------------------
// AttnWindowPool for MI455X (gfx1250, wave32, WMMA bf16 -> f32 accum)
//   out = softmax_window(q . (e@Wk)) pooled over (e@Wv), then @ Wo + bo
// GEMMs: LDS-staged, double-buffered via global_load_async_to_lds_b128
//        (ASYNCcnt) + s_wait_asynccnt, fragments read via ds_load_b128.
//        Pipeline unrolled x2 (K % 64 == 0) -> branch-free steady state,
//        LDS buffer bases are compile-time -> immediate ds offsets.
// ---------------------------------------------------------------------------

typedef __attribute__((ext_vector_type(4)))  __bf16 v4bf;
typedef __attribute__((ext_vector_type(8)))  __bf16 v8bf;
typedef __attribute__((ext_vector_type(16))) __bf16 v16bf;
typedef __attribute__((ext_vector_type(8)))  float  v8f;

#define L_SEQ 16384
#define D_IN  2048
#define D_OUT 2048
#define HEADS 16
#define D_HEAD 128

// LDS staging geometry (per K-step tile: A 64x32 bf16, B 256x32 bf16)
#define ROW_STRIDE  80      // padded: 20 dwords -> conflict-free b128 reads
#define A_REGION    (64 * ROW_STRIDE)            // 5120 B
#define B_REGION    (256 * ROW_STRIDE)           // 20480 B
#define BUF_BYTES   (A_REGION + B_REGION)        // 25600 B

// 16B global->LDS async copy (gfx1250; tracked by ASYNCcnt)
__device__ __forceinline__ void async_copy16(unsigned lds_off,
                                             unsigned long long gaddr) {
  asm volatile("global_load_async_to_lds_b128 %0, %1, off"
               :: "v"(lds_off), "v"(gaddr) : "memory");
}

// ---------------- pack: f32 -> bf16 (elementwise, vec4) ----------------
__global__ __launch_bounds__(256) void pack_bf16_kernel(
    const float* __restrict__ in, __bf16* __restrict__ out, long n4) {
  long i = (long)blockIdx.x * blockDim.x + threadIdx.x;
  if (i >= n4) return;
  float4 v = ((const float4*)in)[i];
  v4bf o;
  o[0] = (__bf16)v.x; o[1] = (__bf16)v.y; o[2] = (__bf16)v.z; o[3] = (__bf16)v.w;
  ((v4bf*)out)[i] = o;
}

// ---------------- pack: f32 [K][N] -> bf16 transposed [N][K] ----------------
__global__ __launch_bounds__(256) void pack_transpose_kernel(
    const float* __restrict__ in, __bf16* __restrict__ out, int K, int N) {
  __shared__ float tile[32][33];
  const int tx = threadIdx.x, ty = threadIdx.y;     // block (32, 8)
  const int n0 = blockIdx.x * 32, k0 = blockIdx.y * 32;
#pragma unroll
  for (int j = 0; j < 32; j += 8)
    tile[ty + j][tx] = in[(long)(k0 + ty + j) * N + (n0 + tx)];
  __syncthreads();
#pragma unroll
  for (int j = 0; j < 32; j += 8)
    out[(long)(n0 + ty + j) * K + (k0 + tx)] = (__bf16)tile[tx][ty + j];
}

// ---------------- WMMA bf16 GEMM: C[M,N] = A[M,K] * BT[N,K]^T ----------------
// block tile 64(M) x 256(N); 8 waves, each wave 32x64 -> 2x4 C frags.
// Requires K % 64 == 0 (pipeline unrolled by 2).
template <bool BF16OUT>
__global__ __launch_bounds__(256) void gemm_bf16_kernel(
    const __bf16* __restrict__ A,   // [M][K] row-major bf16
    const __bf16* __restrict__ BT,  // [N][K] row-major bf16 (B transposed)
    void* __restrict__ outPtr,      // bf16 if BF16OUT else f32
    const float* __restrict__ bias, // used when !BF16OUT (may be null)
    int M, int N, int K) {
  __shared__ __align__(16) unsigned char lds[2][BUF_BYTES];

  const int tid   = threadIdx.x;
  const int lane  = tid & 31;
  const int wave  = tid >> 5;
  const int waveM = wave & 1;
  const int waveN = wave >> 1;
  const int al = lane & 15;   // row/col within fragment
  const int ah = lane >> 4;   // half-wave selector

  const long M0blk = (long)blockIdx.y * 64;
  const long N0blk = (long)blockIdx.x * 256;
  const unsigned base0 = (unsigned)(size_t)(&lds[0][0]);
  const unsigned base1 = (unsigned)(size_t)(&lds[1][0]);

  v8f c[2][4];
#pragma unroll
  for (int i = 0; i < 2; ++i)
#pragma unroll
    for (int j = 0; j < 4; ++j) c[i][j] = (v8f)(0.0f);

  const int NSTEPS = K >> 5;  // even (K % 64 == 0)

  // cooperative async issue of one K-step tile into buffer `base`
  auto issue = [&](int s, unsigned base) __attribute__((always_inline)) {
    const int k0 = s << 5;
    {  // A: 64 rows x 64B = 256 chunks, 1 per thread
      const int row = tid >> 2, sub = tid & 3;
      const unsigned long long g = (unsigned long long)(size_t)(
          A + (size_t)(M0blk + row) * K + k0 + sub * 8);
      async_copy16(base + row * ROW_STRIDE + sub * 16, g);
    }
#pragma unroll
    for (int i = 0; i < 4; ++i) {  // B: 256 rows x 64B = 1024 chunks
      const int ch = tid + i * 256;
      const int row = ch >> 2, sub = ch & 3;
      const unsigned long long g = (unsigned long long)(size_t)(
          BT + (size_t)(N0blk + row) * K + k0 + sub * 8);
      async_copy16(base + A_REGION + row * ROW_STRIDE + sub * 16, g);
    }
  };

  // consume one staged K-step tile: 12 ds_load_b128 + 8 WMMA
  auto compute = [&](const unsigned char* buf) __attribute__((always_inline)) {
    v16bf a[2], b[4];
#pragma unroll
    for (int i = 0; i < 2; ++i) {
      const unsigned char* r = buf + (waveM * 32 + i * 16 + al) * ROW_STRIDE;
      v8bf lo = *(const v8bf*)(r + ah * 16);
      v8bf hi = *(const v8bf*)(r + ah * 16 + 32);
#pragma unroll
      for (int e = 0; e < 8; ++e) { a[i][e] = lo[e]; a[i][e + 8] = hi[e]; }
    }
#pragma unroll
    for (int j = 0; j < 4; ++j) {
      const unsigned char* r =
          buf + A_REGION + (waveN * 64 + j * 16 + al) * ROW_STRIDE;
      v8bf lo = *(const v8bf*)(r + ah * 16);
      v8bf hi = *(const v8bf*)(r + ah * 16 + 32);
#pragma unroll
      for (int e = 0; e < 8; ++e) { b[j][e] = lo[e]; b[j][e + 8] = hi[e]; }
    }
#pragma unroll
    for (int i = 0; i < 2; ++i)
#pragma unroll
      for (int j = 0; j < 4; ++j)
        c[i][j] = __builtin_amdgcn_wmma_f32_16x16x32_bf16(
            false, a[i], false, b[j], (short)0, c[i][j], false, false);
  };

  issue(0, base0);

  // steady state: branch-free double-buffered pipeline, 2 K-steps per trip
  int s = 0;
  for (; s < NSTEPS - 2; s += 2) {
    issue(s + 1, base1);
    asm volatile("s_wait_asynccnt 0x5" ::: "memory");
    __syncthreads();
    compute(lds[0]);
    __syncthreads();
    issue(s + 2, base0);
    asm volatile("s_wait_asynccnt 0x5" ::: "memory");
    __syncthreads();
    compute(lds[1]);
    __syncthreads();
  }
  // tail: steps NSTEPS-2 (in buf0) and NSTEPS-1 (into buf1)
  issue(NSTEPS - 1, base1);
  asm volatile("s_wait_asynccnt 0x5" ::: "memory");
  __syncthreads();
  compute(lds[0]);
  __syncthreads();
  asm volatile("s_wait_asynccnt 0x0" ::: "memory");
  __syncthreads();
  compute(lds[1]);

  // C frag layout: VGPR r, lane -> row = r + ah*8, col = al
#pragma unroll
  for (int i = 0; i < 2; ++i)
#pragma unroll
    for (int j = 0; j < 4; ++j) {
      const long col = N0blk + waveN * 64 + j * 16 + al;
      const long rb  = M0blk + waveM * 32 + i * 16 + ah * 8;
      if (BF16OUT) {
        __bf16* ob = (__bf16*)outPtr;
#pragma unroll
        for (int r = 0; r < 8; ++r)
          ob[(rb + r) * N + col] = (__bf16)c[i][j][r];
      } else {
        float* of = (float*)outPtr;
        const float bv = bias ? bias[col] : 0.0f;
#pragma unroll
        for (int r = 0; r < 8; ++r)
          of[(rb + r) * N + col] = c[i][j][r] + bv;
      }
    }
}

// ---------------- sliding-window attention pool (W_LEFT=3, W_RIGHT=0) -------
// one wave per (t, head); lane owns d = lane*4 .. lane*4+3 of d_head=128.
__global__ __launch_bounds__(256) void attn_window_kernel(
    const __bf16* __restrict__ Kb,  // [L][D_OUT] bf16
    const __bf16* __restrict__ Vb,  // [L][D_OUT] bf16
    const float*  __restrict__ q,   // [HEADS][D_HEAD] f32
    __bf16* __restrict__ pooled,    // [L][D_OUT] bf16
    int L) {
  const int lane = threadIdx.x & 31;
  const int wave = threadIdx.x >> 5;
  const int pair = blockIdx.x * 8 + wave;  // t*HEADS + h
  const int t = pair >> 4;
  const int h = pair & 15;
  if (t >= L) return;

  const int dbase = h * D_HEAD + lane * 4;
  const float4 qv = *(const float4*)(q + dbase);

  float att[4];
  float vval[4][4];
#pragma unroll
  for (int w = 0; w < 4; ++w) {
    const int tp = t - 3 + w;       // window offsets -3..0 (uniform per wave)
    if (tp >= 0) {
      const v4bf kv = *(const v4bf*)(Kb + (long)tp * D_OUT + dbase);
      const v4bf vv = *(const v4bf*)(Vb + (long)tp * D_OUT + dbase);
      float s = qv.x * (float)kv[0] + qv.y * (float)kv[1] +
                qv.z * (float)kv[2] + qv.w * (float)kv[3];
#pragma unroll
      for (int off = 16; off > 0; off >>= 1) s += __shfl_xor(s, off, 32);
      att[w] = s * 0.08838834764831845f;  // 1/sqrt(128)
#pragma unroll
      for (int e = 0; e < 4; ++e) vval[w][e] = (float)vv[e];
    } else {
      att[w] = -1e30f;
#pragma unroll
      for (int e = 0; e < 4; ++e) vval[w][e] = 0.0f;
    }
  }

  float m = att[0];
#pragma unroll
  for (int w = 1; w < 4; ++w) m = fmaxf(m, att[w]);
  float p[4], sum = 0.0f;
#pragma unroll
  for (int w = 0; w < 4; ++w) { p[w] = __expf(att[w] - m); sum += p[w]; }
  const float inv = 1.0f / sum;

  v4bf o;
#pragma unroll
  for (int e = 0; e < 4; ++e) {
    float acc = 0.0f;
#pragma unroll
    for (int w = 0; w < 4; ++w) acc += p[w] * vval[w][e];
    o[e] = (__bf16)(acc * inv);
  }
  *(v4bf*)(pooled + (long)t * D_OUT + dbase) = o;
}

// ---------------------------------------------------------------------------
extern "C" void kernel_launch(void* const* d_in, const int* in_sizes, int n_in,
                              void* d_out, int out_size, void* d_ws, size_t ws_size,
                              hipStream_t stream) {
  const float* e  = (const float*)d_in[0];  // [L][D_IN]
  const float* q  = (const float*)d_in[1];  // [HEADS][D_HEAD]
  const float* Wk = (const float*)d_in[2];  // [D_IN][D_OUT]
  const float* Wv = (const float*)d_in[3];
  const float* Wo = (const float*)d_in[4];  // [D_OUT][D_OUT]
  const float* bo = (const float*)d_in[5];  // [D_OUT]
  float* out = (float*)d_out;               // [L][D_OUT] f32

  char* ws = (char*)d_ws;
  __bf16* eB  = (__bf16*)ws; ws += (size_t)L_SEQ * D_IN * 2;
  __bf16* WkT = (__bf16*)ws; ws += (size_t)D_IN * D_OUT * 2;
  __bf16* WvT = (__bf16*)ws; ws += (size_t)D_IN * D_OUT * 2;
  __bf16* WoT = (__bf16*)ws; ws += (size_t)D_OUT * D_OUT * 2;
  __bf16* kB  = (__bf16*)ws; ws += (size_t)L_SEQ * D_OUT * 2;
  __bf16* vB  = (__bf16*)ws; ws += (size_t)L_SEQ * D_OUT * 2;
  __bf16* pB  = (__bf16*)ws; ws += (size_t)L_SEQ * D_OUT * 2;

  // 1) packs
  {
    const long n4 = (long)L_SEQ * D_IN / 4;
    pack_bf16_kernel<<<(int)((n4 + 255) / 256), 256, 0, stream>>>(e, eB, n4);
  }
  {
    dim3 tb(32, 8);
    dim3 tgK(D_OUT / 32, D_IN / 32);
    pack_transpose_kernel<<<tgK, tb, 0, stream>>>(Wk, WkT, D_IN, D_OUT);
    pack_transpose_kernel<<<tgK, tb, 0, stream>>>(Wv, WvT, D_IN, D_OUT);
    dim3 tgO(D_OUT / 32, D_OUT / 32);
    pack_transpose_kernel<<<tgO, tb, 0, stream>>>(Wo, WoT, D_OUT, D_OUT);
  }

  // 2) K = e@Wk, V = e@Wv (bf16 outputs)
  dim3 gg(D_OUT / 256, L_SEQ / 64);
  gemm_bf16_kernel<true><<<gg, 256, 0, stream>>>(eB, WkT, (void*)kB, nullptr,
                                                 L_SEQ, D_OUT, D_IN);
  gemm_bf16_kernel<true><<<gg, 256, 0, stream>>>(eB, WvT, (void*)vB, nullptr,
                                                 L_SEQ, D_OUT, D_IN);

  // 3) sliding-window softmax pool
  attn_window_kernel<<<L_SEQ * HEADS / 8, 256, 0, stream>>>(kB, vB, q, pB, L_SEQ);

  // 4) out = pooled @ Wo + bo (f32 output)
  gemm_bf16_kernel<false><<<gg, 256, 0, stream>>>(pB, WoT, (void*)out, bo,
                                                  L_SEQ, D_OUT, D_OUT);
}
